// Transformer_6717328851523
// MI455X (gfx1250) — matrix operand order repaired and checked by reading, not compiled, not run on hardware
//
#include <hip/hip_runtime.h>
#include <cstddef>

// ---------- types ----------
typedef __attribute__((ext_vector_type(16))) _Float16 v16h;
typedef __attribute__((ext_vector_type(8)))  _Float16 v8h;
typedef __attribute__((ext_vector_type(8)))  float    v8f;
typedef __attribute__((ext_vector_type(4)))  float    v4f;

#define DEV static __device__ __forceinline__

static constexpr int Bb  = 8;
static constexpr int Ss  = 1024;
static constexpr int Dd  = 1024;
static constexpr int Hh  = 16;
static constexpr int HD  = 64;
static constexpr int FF  = 4096;
static constexpr int OUT = 1000;
static constexpr int Ll  = 2;

// ---------- CDNA5 async global->LDS (ASYNCcnt) path, guarded ----------
#if __has_builtin(__builtin_amdgcn_global_load_async_to_lds_b128) && \
    __has_builtin(__builtin_amdgcn_s_wait_asynccnt)
#define USE_ASYNC_LDS 1
#else
#define USE_ASYNC_LDS 0
#endif

#if USE_ASYNC_LDS
#define AS_GLOBAL __attribute__((address_space(1)))
#define AS_LDS    __attribute__((address_space(3)))
typedef int async16_t __attribute__((vector_size(16)));  // builtin's exact pointee type
// one 16-byte chunk, per-lane, global -> LDS without touching VGPRs
DEV void stage16(const _Float16* g, _Float16* l) {
  __builtin_amdgcn_global_load_async_to_lds_b128(
      (AS_GLOBAL async16_t*)g, (AS_LDS async16_t*)l, /*imm offset*/ 0, /*cpol*/ 0);
}
DEV void stage_wait() { __builtin_amdgcn_s_wait_asynccnt(0); }
#else
DEV void stage16(const _Float16* g, _Float16* l) { *(v8h*)l = *(const v8h*)g; }
DEV void stage_wait() {}
#endif

DEV v8f zero8() {
  v8f z;
  #pragma unroll
  for (int i = 0; i < 8; ++i) z[i] = 0.0f;
  return z;
}

DEV v8f wmma_f16(v16h a, v16h b, v8f c) {
  // D = A(16x32 f16) * B(32x16 f16) + C(16x16 f32)
  return __builtin_amdgcn_wmma_f32_16x16x32_f16(
      false, a, false, b, (short)0, c, false, false);
}

// A fragment: 16x32 f16, row-major tile base with pitch (halfs).
// lane: m = lane&15 ; halves 0..7 = A[m][kb..kb+7], halves 8..15 = A[m][kb+16..kb+23],
// kb = k0 + (lane>>4)*8  (ISA 7.12.2, 16-bit A 16x32)
DEV v16h load_a_frag(const _Float16* __restrict__ base, int pitch, int m0, int k0, int lane) {
  const _Float16* p = base + (size_t)(m0 + (lane & 15)) * pitch + k0 + ((lane >> 4) << 3);
  v8h lo = *(const v8h*)p;
  v8h hi = *(const v8h*)(p + 16);
  v16h r;
  #pragma unroll
  for (int i = 0; i < 8; ++i) { r[i] = lo[i]; r[i + 8] = hi[i]; }
  return r;
}

// B fragment: 32x16 f16 from an N-major staging tile Bs[n][k] (Bs[n][k] == B[k][n]).
// lane: n = n0 + (lane&15); halves 0..7 = Bs[n][kb..kb+7], halves 8..15 = Bs[n][kb+16..kb+23], kb = k0 + (lane>>4)*8
DEV v16h load_b_frag(const _Float16* __restrict__ base, int pitch, int n0, int k0, int lane) {
  const _Float16* p = base + (size_t)(n0 + (lane & 15)) * pitch + k0 + ((lane >> 4) << 3);
  v8h lo = *(const v8h*)p;
  v8h hi = *(const v8h*)(p + 16);
  v16h r;
  #pragma unroll
  for (int i = 0; i < 8; ++i) { r[i] = lo[i]; r[i + 8] = hi[i]; }
  return r;
}

// ---------- f32 -> f16 conversion ----------
__global__ void cvt4_kernel(const float* __restrict__ in, _Float16* __restrict__ out, int n4) {
  int i = blockIdx.x * 256 + threadIdx.x;
  if (i < n4) {
    v4f v = ((const v4f*)in)[i];
    #pragma unroll
    for (int j = 0; j < 4; ++j) out[4 * i + j] = (_Float16)v[j];
  }
}

// ---------- layernorm: f32 row -> f16 row ----------
__global__ __launch_bounds__(256)
void ln_kernel(const float* __restrict__ x, const float* __restrict__ g,
               const float* __restrict__ bt, _Float16* __restrict__ out) {
  __shared__ float red[256];
  const int row = blockIdx.x, t = threadIdx.x;
  const float* xr = x + (size_t)row * Dd;
  v4f v = *(const v4f*)(xr + t * 4);
  float s  = v[0] + v[1] + v[2] + v[3];
  float s2 = v[0]*v[0] + v[1]*v[1] + v[2]*v[2] + v[3]*v[3];
  red[t] = s; __syncthreads();
  #pragma unroll
  for (int o = 128; o > 0; o >>= 1) { if (t < o) red[t] += red[t + o]; __syncthreads(); }
  const float mean = red[0] * (1.0f / Dd);
  __syncthreads();
  red[t] = s2; __syncthreads();
  #pragma unroll
  for (int o = 128; o > 0; o >>= 1) { if (t < o) red[t] += red[t + o]; __syncthreads(); }
  const float var  = red[0] * (1.0f / Dd) - mean * mean;
  const float rinv = rsqrtf(var + 1e-5f);
  _Float16* orow = out + (size_t)row * Dd;
  #pragma unroll
  for (int j = 0; j < 4; ++j) {
    const int d = t * 4 + j;
    orow[d] = (_Float16)((v[j] - mean) * rinv * g[d] + bt[d]);
  }
}

// ---------- WMMA GEMM: C[M,N] = A[M,K](f16) * B[K,N](f16) + bias (+gelu) (+resid) ----------
// block tile 128x128, 256 threads = 8 waves (2 M x 4 N), wave tile 64x32 (4x2 frags)
template <int BIAS, int GELU, int RESID, int OUTF16>
__global__ __launch_bounds__(256)
void gemm_wmma_kernel(const _Float16* __restrict__ A, int lda,
                      const _Float16* __restrict__ B, int ldb,
                      const float* __restrict__ bias,
                      const float* __restrict__ resid, int ldr,
                      float* __restrict__ Cf, _Float16* __restrict__ Ch, int ldc,
                      int M, int N, int K) {
  constexpr int KT = 32;
  constexpr int AP = KT + 8;   // padded pitch (halfs)
  constexpr int BP = KT + 8;
  __shared__ _Float16 As[2][128][AP];
  __shared__ _Float16 Bs[2][128][BP];   // N-major: Bs[n][k]

  const int t = threadIdx.x, wave = t >> 5, lane = t & 31;
  const int m0 = (wave >> 2) * 64;      // wave M origin in block tile
  const int n0 = (wave & 3) * 32;       // wave N origin in block tile
  const int bm = blockIdx.y * 128, bn = blockIdx.x * 128;

  v8f acc[4][2];
  #pragma unroll
  for (int mf = 0; mf < 4; ++mf)
    #pragma unroll
    for (int nf = 0; nf < 2; ++nf) acc[mf][nf] = zero8();

  auto load_stage = [&](int s, int kk) {
    // A tile: 128 x 32 halfs; 512 chunks of 8 halfs; 2 per thread.
    // No transpose needed -> CDNA5 async global->LDS when available.
    #pragma unroll
    for (int i = 0; i < 2; ++i) {
      const int c = t + 256 * i;
      const int row = c >> 2, col = (c & 3) << 3;
      stage16(A + (size_t)(bm + row) * lda + kk + col, &As[s][row][col]);
    }
    // B tile: 32 x 128 halfs, transposed into Bs[n][k] (VGPR path: scatter stores)
    #pragma unroll
    for (int i = 0; i < 2; ++i) {
      const int c = t + 256 * i;
      const int krow = c >> 4, ncol = (c & 15) << 3;
      v8h d = *(const v8h*)(B + (size_t)(kk + krow) * ldb + bn + ncol);
      #pragma unroll
      for (int j = 0; j < 8; ++j) Bs[s][ncol + j][krow] = d[j];
    }
  };

  const int nkt = K / KT;
  load_stage(0, 0);
  for (int kt = 0; kt < nkt; ++kt) {
    stage_wait();          // my async chunks for the stage about to be consumed
    __syncthreads();       // everyone's stage writes visible
    if (kt + 1 < nkt) {
      // soft prefetch of the tile after next (codegen: global_prefetch_b8)
      if (kt + 2 < nkt)
        __builtin_prefetch(A + (size_t)(bm + (t >> 1)) * lda + (kt + 2) * KT, 0, 1);
      load_stage((kt + 1) & 1, (kt + 1) * KT);
    }
    const int s = kt & 1;
    v16h af[4];
    #pragma unroll
    for (int mf = 0; mf < 4; ++mf)
      af[mf] = load_a_frag(&As[s][0][0], AP, m0 + mf * 16, 0, lane);
    v16h bf[2];
    #pragma unroll
    for (int nf = 0; nf < 2; ++nf)
      bf[nf] = load_b_frag(&Bs[s][0][0], BP, n0 + nf * 16, 0, lane);
    #pragma unroll
    for (int mf = 0; mf < 4; ++mf)
      #pragma unroll
      for (int nf = 0; nf < 2; ++nf)
        acc[mf][nf] = wmma_f16(af[mf], bf[nf], acc[mf][nf]);
  }

  // epilogue (C layout: vgpr r -> row r + 8*(lane>>4); col = lane&15)
  const int rbase = bm + m0, cbase = bn + n0;
  #pragma unroll
  for (int mf = 0; mf < 4; ++mf)
    #pragma unroll
    for (int nf = 0; nf < 2; ++nf) {
      const int col = cbase + nf * 16 + (lane & 15);
      const float bv = BIAS ? bias[col] : 0.0f;
      #pragma unroll
      for (int r = 0; r < 8; ++r) {
        const int row = rbase + mf * 16 + r + ((lane >> 4) << 3);
        float v = acc[mf][nf][r] + bv;
        if (GELU)  v = 0.5f * v * (1.0f + erff(v * 0.70710678118f));
        if (RESID) v += resid[(size_t)row * ldr + col];
        if (OUTF16) Ch[(size_t)row * ldc + col] = (_Float16)v;
        else        Cf[(size_t)row * ldc + col] = v;
      }
    }
}

// ---------- flash attention ----------
// grid: (S/64, H, B); block: 128 threads = 4 waves, wave owns 16 q rows.
// qkv f16 layout: row = b*S+s (pitch 3D); q at h*64, k at D+h*64, v at 2D+h*64
__global__ __launch_bounds__(128)
void attn_kernel(const _Float16* __restrict__ qkv, _Float16* __restrict__ o16) {
  __shared__ _Float16 Kt[64][72];       // K tile, row-major [krow][hd]
  __shared__ _Float16 Vt[64][72];       // V tile, transposed [hd][krow]
  __shared__ _Float16 Ps[4][16][72];    // per-wave P staging (16 x 64)

  const int t = threadIdx.x, wave = t >> 5, lane = t & 31;
  const int b = blockIdx.z, h = blockIdx.y, qt = blockIdx.x;
  const int QKVP = 3 * Dd;
  const float scale = 0.125f;           // 1/sqrt(64)

  const _Float16* qp = qkv + (size_t)(b * Ss + qt * 64 + wave * 16) * QKVP + h * HD;
  const v16h qa0 = load_a_frag(qp, QKVP, 0, 0, lane);
  const v16h qa1 = load_a_frag(qp, QKVP, 0, 32, lane);

  float mrow[8], lrow[8];
  v8f of[4];
  #pragma unroll
  for (int r = 0; r < 8; ++r) { mrow[r] = -1e30f; lrow[r] = 0.0f; }
  #pragma unroll
  for (int nf = 0; nf < 4; ++nf) of[nf] = zero8();

  const _Float16* kb = qkv + (size_t)(b * Ss) * QKVP + Dd + h * HD;
  const _Float16* vb = qkv + (size_t)(b * Ss) * QKVP + 2 * Dd + h * HD;

  for (int kv = 0; kv < Ss; kv += 64) {
    __syncthreads();
    // K tile row-major (async global->LDS); V tile transposed (VGPR path)
    #pragma unroll
    for (int i = 0; i < 4; ++i) {
      const int c = t + 128 * i;
      const int row = c >> 3, col = (c & 7) << 3;
      stage16(kb + (size_t)(kv + row) * QKVP + col, &Kt[row][col]);
      v8h dv = *(const v8h*)(vb + (size_t)(kv + row) * QKVP + col);
      #pragma unroll
      for (int j = 0; j < 8; ++j) Vt[col + j][row] = dv[j];
    }
    stage_wait();
    __syncthreads();

    // S = Q K^T  (16 x 64 per wave)
    v8f sf[4];
    #pragma unroll
    for (int nf = 0; nf < 4; ++nf) {
      v8f z = zero8();
      z = wmma_f16(qa0, load_b_frag(&Kt[0][0], 72, nf * 16, 0, lane), z);
      z = wmma_f16(qa1, load_b_frag(&Kt[0][0], 72, nf * 16, 32, lane), z);
      sf[nf] = z;
    }

    // online softmax: per-register row stats, reduce across the 16-lane half
    float mnew[8], alpha[8];
    #pragma unroll
    for (int r = 0; r < 8; ++r) {
      float mx = -1e30f;
      #pragma unroll
      for (int nf = 0; nf < 4; ++nf) mx = fmaxf(mx, sf[nf][r]);
      #pragma unroll
      for (int off = 1; off < 16; off <<= 1) mx = fmaxf(mx, __shfl_xor(mx, off, 32));
      mx *= scale;
      mnew[r]  = fmaxf(mrow[r], mx);
      alpha[r] = __expf(mrow[r] - mnew[r]);
      mrow[r]  = mnew[r];
    }

    float rsum[8];
    #pragma unroll
    for (int r = 0; r < 8; ++r) rsum[r] = 0.0f;
    #pragma unroll
    for (int nf = 0; nf < 4; ++nf) {
      #pragma unroll
      for (int r = 0; r < 8; ++r) {
        const float p = __expf(sf[nf][r] * scale - mnew[r]);
        rsum[r] += p;
        Ps[wave][r + ((lane >> 4) << 3)][nf * 16 + (lane & 15)] = (_Float16)p;
      }
    }
    #pragma unroll
    for (int r = 0; r < 8; ++r) {
      float s = rsum[r];
      #pragma unroll
      for (int off = 1; off < 16; off <<= 1) s += __shfl_xor(s, off, 32);
      lrow[r] = lrow[r] * alpha[r] + s;
    }
    #pragma unroll
    for (int nf = 0; nf < 4; ++nf) {
      #pragma unroll
      for (int r = 0; r < 8; ++r) of[nf][r] *= alpha[r];
    }

    // O += P V   (DS ops are in-order within a wave: Ps RAW is safe)
    #pragma unroll
    for (int nf = 0; nf < 4; ++nf) {
      of[nf] = wmma_f16(load_a_frag(&Ps[wave][0][0], 72, 0, 0, lane),
                        load_b_frag(&Vt[0][0], 72, nf * 16, 0, lane), of[nf]);
      of[nf] = wmma_f16(load_a_frag(&Ps[wave][0][0], 72, 0, 32, lane),
                        load_b_frag(&Vt[0][0], 72, nf * 16, 32, lane), of[nf]);
    }
  }

  _Float16* orow = o16 + (size_t)(b * Ss + qt * 64 + wave * 16) * Dd + h * HD;
  #pragma unroll
  for (int nf = 0; nf < 4; ++nf) {
    const int col = nf * 16 + (lane & 15);
    #pragma unroll
    for (int r = 0; r < 8; ++r) {
      const int row = r + ((lane >> 4) << 3);
      orow[(size_t)row * Dd + col] = (_Float16)(of[nf][r] / lrow[r]);
    }
  }
}

// ---------- mean pool over S ----------
__global__ void pool_kernel(const float* __restrict__ x, float* __restrict__ pooled) {
  const int d = blockIdx.x * 256 + threadIdx.x;
  const int b = blockIdx.y;
  float s = 0.0f;
  for (int si = 0; si < Ss; ++si) s += x[((size_t)b * Ss + si) * Dd + d];
  pooled[b * Dd + d] = s * (1.0f / Ss);
}

// ---------- classification head (tiny) ----------
__global__ void head_kernel(const float* __restrict__ pooled, const float* __restrict__ w,
                            const float* __restrict__ bias, float* __restrict__ out) {
  const int o = blockIdx.x * 256 + threadIdx.x;
  if (o >= OUT) return;
  const int b = blockIdx.y;
  const float* p = pooled + b * Dd;
  float acc = bias[o];
  for (int d = 0; d < Dd; ++d) acc += p[d] * w[(size_t)d * OUT + o];
  out[b * OUT + o] = acc;
}

// ---------- host orchestration ----------
extern "C" void kernel_launch(void* const* d_in, const int* in_sizes, int n_in,
                              void* d_out, int out_size, void* d_ws, size_t ws_size,
                              hipStream_t stream) {
  (void)in_sizes; (void)n_in; (void)out_size; (void)ws_size;
  const float* x          = (const float*)d_in[0];
  const float* qkv_w      = (const float*)d_in[1];
  const float* qkv_b      = (const float*)d_in[2];
  const float* attn_out_w = (const float*)d_in[3];
  const float* attn_out_b = (const float*)d_in[4];
  const float* ln1_g      = (const float*)d_in[5];
  const float* ln1_b      = (const float*)d_in[6];
  const float* ln2_g      = (const float*)d_in[7];
  const float* ln2_b      = (const float*)d_in[8];
  const float* ff_w1      = (const float*)d_in[9];
  const float* ff_b1      = (const float*)d_in[10];
  const float* ff_w2      = (const float*)d_in[11];
  const float* ff_b2      = (const float*)d_in[12];
  const float* fc_w       = (const float*)d_in[13];
  const float* fc_b       = (const float*)d_in[14];
  float* out = (float*)d_out;

  const int ROWS = Bb * Ss;  // 8192
  char* ws = (char*)d_ws;
  size_t off = 0;
  auto alloc = [&](size_t bytes) { char* p = ws + off; off = (off + bytes + 255) & ~(size_t)255; return p; };

  _Float16* qkvw16 = (_Float16*)alloc((size_t)Ll * Dd * 3 * Dd * 2);
  _Float16* aow16  = (_Float16*)alloc((size_t)Ll * Dd * Dd * 2);
  _Float16* ffw1h  = (_Float16*)alloc((size_t)Dd * FF * 2);
  _Float16* ffw2h  = (_Float16*)alloc((size_t)FF * Dd * 2);
  float*    xbuf   = (float*)   alloc((size_t)ROWS * Dd * 4);
  _Float16* h16    = (_Float16*)alloc((size_t)ROWS * Dd * 2);
  _Float16* qkv16  = (_Float16*)alloc((size_t)ROWS * 3 * Dd * 2);
  _Float16* o16    = (_Float16*)alloc((size_t)ROWS * Dd * 2);
  _Float16* g16    = (_Float16*)alloc((size_t)ROWS * FF * 2);
  float*    pooled = (float*)   alloc((size_t)Bb * Dd * 4);

  // residual stream starts as x
  (void)hipMemcpyAsync(xbuf, x, (size_t)ROWS * Dd * sizeof(float),
                       hipMemcpyDeviceToDevice, stream);

  // convert weights to f16
  auto cvt = [&](const float* src, _Float16* dst, size_t n) {
    int n4 = (int)(n / 4);
    cvt4_kernel<<<(n4 + 255) / 256, 256, 0, stream>>>(src, dst, n4);
  };
  cvt(qkv_w,      qkvw16, (size_t)Ll * Dd * 3 * Dd);
  cvt(attn_out_w, aow16,  (size_t)Ll * Dd * Dd);
  cvt(ff_w1,      ffw1h,  (size_t)Dd * FF);
  cvt(ff_w2,      ffw2h,  (size_t)FF * Dd);

  for (int l = 0; l < Ll; ++l) {
    // h = LN1(x)
    ln_kernel<<<ROWS, 256, 0, stream>>>(xbuf, ln1_g, ln1_b, h16);
    // qkv = h @ Wqkv + b  (f16 out)
    gemm_wmma_kernel<1, 0, 0, 1><<<dim3(3 * Dd / 128, ROWS / 128), 256, 0, stream>>>(
        h16, Dd, qkvw16 + (size_t)l * Dd * 3 * Dd, 3 * Dd,
        qkv_b + (size_t)l * 3 * Dd, nullptr, 0, nullptr, qkv16, 3 * Dd,
        ROWS, 3 * Dd, Dd);
    // attention
    attn_kernel<<<dim3(Ss / 64, Hh, Bb), 128, 0, stream>>>(qkv16, o16);
    // x = x + o @ Wo + b  (f32 out, fused residual)
    gemm_wmma_kernel<1, 0, 1, 0><<<dim3(Dd / 128, ROWS / 128), 256, 0, stream>>>(
        o16, Dd, aow16 + (size_t)l * Dd * Dd, Dd,
        attn_out_b + (size_t)l * Dd, xbuf, Dd, xbuf, nullptr, Dd,
        ROWS, Dd, Dd);
  }

  // h = LN2(x); g = gelu(h @ W1 + b1); x = x + g @ W2 + b2
  ln_kernel<<<ROWS, 256, 0, stream>>>(xbuf, ln2_g, ln2_b, h16);
  gemm_wmma_kernel<1, 1, 0, 1><<<dim3(FF / 128, ROWS / 128), 256, 0, stream>>>(
      h16, Dd, ffw1h, FF, ff_b1, nullptr, 0, nullptr, g16, FF, ROWS, FF, Dd);
  gemm_wmma_kernel<1, 0, 1, 0><<<dim3(Dd / 128, ROWS / 128), 256, 0, stream>>>(
      g16, FF, ffw2h, Dd, ff_b2, xbuf, Dd, xbuf, nullptr, Dd, ROWS, Dd, FF);

  // pool + head
  pool_kernel<<<dim3(Dd / 256, Bb), 256, 0, stream>>>(xbuf, pooled);
  head_kernel<<<dim3((OUT + 255) / 256, Bb), 256, 0, stream>>>(pooled, fc_w, fc_b, out);
}
